// MoE_37778532335918
// MI455X (gfx1250) — matrix operand order repaired and checked by reading, not compiled, not run on hardware
//
#include <hip/hip_runtime.h>
#include <hip/hip_bf16.h>

// ---- problem constants (from reference) ----
#define NTOK 8192      // B*T = 4*2048
#define CDIM 1024
#define FDIM 2048
#define NEXP 8
#define TOPK 2

// ---- tiling ----
#define BM 128
#define BN 128
#define BK 32
#define MAXROWS (NTOK*TOPK + NEXP*BM)   // 17408 padded gather rows

// ---- workspace layout (bytes) ----
#define WE_OFF   0u                              // float[NTOK*NEXP]
#define GIDX_OFF (WE_OFF + NTOK*NEXP*4u)         // int[MAXROWS]
#define WROW_OFF (GIDX_OFF + MAXROWS*4u)         // float[MAXROWS]
#define OFFS_OFF (WROW_OFF + MAXROWS*4u)         // int[NEXP+1]
#define HID_OFF  (512u*1024u)                    // bf16[MAXROWS*FDIM], blocked
#define HID_BYTES (MAXROWS*FDIM*2u)              // 71,303,168
#define W1B_OFF  (HID_OFF + HID_BYTES)           // bf16 blocked w1
#define W3B_OFF  (W1B_OFF + NEXP*CDIM*FDIM*2u)
#define W2B_OFF  (W3B_OFF + NEXP*CDIM*FDIM*2u)   // end ~172.5 MB

typedef __attribute__((ext_vector_type(16))) __bf16 v16bf;
typedef __attribute__((ext_vector_type(4)))  __bf16 v4bf;
typedef __attribute__((ext_vector_type(8)))  float  v8f;
typedef int v4i_vs __attribute__((vector_size(16)));   // matches builtin param

__device__ __forceinline__ __bf16 f2bf(float f) { return (__bf16)f; }

// ---- CDNA5 async global->LDS copy (16B per lane), ASYNCcnt-tracked --------
#if __has_builtin(__builtin_amdgcn_global_load_async_to_lds_b128)
#define ASYNC_CP16(lds_p, g_p)                                                \
  __builtin_amdgcn_global_load_async_to_lds_b128(                             \
      (__attribute__((address_space(1))) v4i_vs*)(size_t)(g_p),               \
      (__attribute__((address_space(3))) v4i_vs*)(unsigned)(size_t)(lds_p),   \
      0, 0)
#else
#define ASYNC_CP16(lds_p, g_p)                                                \
  asm volatile("global_load_async_to_lds_b128 %0, %1, off" ::                 \
                   "v"((unsigned)(size_t)(lds_p)),                            \
                   "v"((unsigned long long)(size_t)(g_p))                     \
               : "memory")
#endif

#if __has_builtin(__builtin_amdgcn_s_wait_asynccnt)
#define WAIT_ASYNC(n) __builtin_amdgcn_s_wait_asynccnt(n)
#else
#define WAIT_ASYNC(n) asm volatile("s_wait_asynccnt %0" ::"n"(n) : "memory")
#endif

// --- WMMA fragment-layout helpers (CDNA5 CDNA5 ISA VGPR layouts) ----------
// A (16x32 bf16): lane(0-15)=row, elems 0-7 -> K 0-7, elems 8-15 -> K 16-23;
//                 lanes 16-31:   K 8-15 / 24-31.  16-row subtile = 512 elems.
// B (32x16 bf16): lane&15 = column N, same K order as A.
__device__ __forceinline__ int a_swz(int m, int k) {
  int g = k >> 3;
  int ln = ((g & 1) << 4) + (m & 15);
  int es = (k & 7) + ((g >> 1) << 3);
  return ((m >> 4) << 9) + (ln << 4) + es;
}
__device__ __forceinline__ int b_swz(int k, int n) {
  int g = k >> 3;
  int ln = ((g & 1) << 4) + (n & 15);
  int es = (k & 7) + ((g >> 1) << 3);
  return ((n >> 4) << 9) + (ln << 4) + es;
}

// ---------------------------------------------------------------------------
// Kernel 0: stream-convert fp32 weights -> bf16 in fragment-blocked layout:
// dst[((e*(N/128)+nt)*(K/32)+kt)*4096 + b_swz(k&31, n&127)]
// ---------------------------------------------------------------------------
__global__ __launch_bounds__(256)
void wcvt(const float* __restrict__ src, unsigned short* __restrict__ dst_,
          int K, int N) {
  __bf16* dst = (__bf16*)dst_;
  int tid = blockIdx.x * 256 + threadIdx.x;
  int n = tid % N;
  int rest = tid / N;
  int kq = rest % (K >> 2);
  int e = rest / (K >> 2);
  int k = kq << 2;
  const float* s = src + (size_t)(e * K + k) * N + n;     // 4 rows, coalesced
  v4bf v = {f2bf(s[0]), f2bf(s[(size_t)N]), f2bf(s[2 * (size_t)N]),
            f2bf(s[3 * (size_t)N])};
  size_t tile = ((size_t)(e * (N >> 7) + (n >> 7)) * (K >> 5) + (k >> 5)) << 12;
  *(v4bf*)&dst[tile + b_swz(k & 31, n & 127)] = v;        // 8B contiguous
}

// ---------------------------------------------------------------------------
// Kernel 1: router. One wave per token: logits, softmax, top-2, renormalize.
// ---------------------------------------------------------------------------
__global__ __launch_bounds__(256)
void moe_router(const float* __restrict__ x, const float* __restrict__ rw,
                float* __restrict__ we) {
  __shared__ float srw[NEXP * CDIM];
  int tid = threadIdx.x;
  for (int i = tid; i < NEXP * CDIM; i += 256) {
    int c = i >> 3, e = i & 7;
    srw[e * CDIM + c] = rw[i];
  }
  __syncthreads();

  int wid = tid >> 5, lane = tid & 31;
  int t = blockIdx.x * 8 + wid;

  float acc[NEXP];
#pragma unroll
  for (int e = 0; e < NEXP; ++e) acc[e] = 0.f;
  const float* xr = x + (size_t)t * CDIM;
  for (int c = lane; c < CDIM; c += 32) {
    float xv = xr[c];
#pragma unroll
    for (int e = 0; e < NEXP; ++e) acc[e] += xv * srw[e * CDIM + c];
  }
#pragma unroll
  for (int e = 0; e < NEXP; ++e) {
#pragma unroll
    for (int off = 16; off > 0; off >>= 1)
      acc[e] += __shfl_xor(acc[e], off, 32);
  }
  if (lane == 0) {
    float mx = acc[0];
#pragma unroll
    for (int e = 1; e < NEXP; ++e) mx = fmaxf(mx, acc[e]);
    float p[NEXP];
#pragma unroll
    for (int e = 0; e < NEXP; ++e) p[e] = __expf(acc[e] - mx);
    int i1 = 0;
#pragma unroll
    for (int e = 1; e < NEXP; ++e) if (p[e] > p[i1]) i1 = e;
    int i2 = (i1 == 0) ? 1 : 0;
#pragma unroll
    for (int e = 0; e < NEXP; ++e) if (e != i1 && p[e] > p[i2]) i2 = e;
    float inv = 1.f / (p[i1] + p[i2]);
#pragma unroll
    for (int e = 0; e < NEXP; ++e)
      we[(size_t)t * NEXP + e] =
          (e == i1) ? p[i1] * inv : ((e == i2) ? p[i2] * inv : 0.f);
  }
}

// ---------------------------------------------------------------------------
// Kernel 2: per-expert compacted token lists, BM-rounded scan, -1 padding.
// ---------------------------------------------------------------------------
__global__ __launch_bounds__(256)
void moe_gather(const float* __restrict__ we, int* __restrict__ gidx,
                float* __restrict__ wrow, int* __restrict__ offs) {
  __shared__ int cnt[NEXP], base[NEXP], fill[NEXP];
  __shared__ int total;
  int tid = threadIdx.x;
  if (tid < NEXP) cnt[tid] = 0;
  __syncthreads();
  for (int t = tid; t < NTOK; t += 256) {
#pragma unroll
    for (int e = 0; e < NEXP; ++e)
      if (we[(size_t)t * NEXP + e] > 0.f) atomicAdd(&cnt[e], 1);
  }
  __syncthreads();
  if (tid == 0) {
    int run = 0;
    for (int e = 0; e < NEXP; ++e) {
      base[e] = run;
      offs[e] = run;
      run += ((cnt[e] + BM - 1) / BM) * BM;
    }
    offs[NEXP] = run;
    total = run;
  }
  __syncthreads();
  for (int i = tid; i < total; i += 256) { gidx[i] = -1; wrow[i] = 0.f; }
  if (tid < NEXP) fill[tid] = 0;
  __syncthreads();
  for (int t = tid; t < NTOK; t += 256) {
#pragma unroll
    for (int e = 0; e < NEXP; ++e) {
      float w = we[(size_t)t * NEXP + e];
      if (w > 0.f) {
        int p = atomicAdd(&fill[e], 1);
        gidx[base[e] + p] = t;
        wrow[base[e] + p] = w;
      }
    }
  }
}

// ---------------------------------------------------------------------------
// Kernel 3: hidden = silu(Xg @ w1) * (Xg @ w3) -> blocked bf16.
// B tiles stream via async global->LDS; A (gathered x rows) is branch-free,
// register-pipelined one k-step ahead. Double-buffered LDS, 8 waves (2x4).
// ---------------------------------------------------------------------------
__global__ __launch_bounds__(256)
void moe_ffn1(const float* __restrict__ x,
              const unsigned short* __restrict__ w1b_,
              const unsigned short* __restrict__ w3b_,
              const int* __restrict__ gidx, const int* __restrict__ offs,
              unsigned short* __restrict__ hid_) {
  int e = blockIdx.z;
  int rowBase = offs[e];
  int rows = offs[e + 1] - rowBase;
  int mt = blockIdx.x;
  if (mt * BM >= rows) return;
  int nt = blockIdx.y;
  int f0 = nt * BN;

  __bf16* hidden = (__bf16*)hid_;
  const __bf16* w1b =
      (const __bf16*)w1b_ + (((size_t)(e * (FDIM / BN) + nt) * (CDIM / BK)) << 12);
  const __bf16* w3b =
      (const __bf16*)w3b_ + (((size_t)(e * (FDIM / BN) + nt) * (CDIM / BK)) << 12);

  __shared__ int stok[BM];
  __shared__ __align__(16) __bf16 sA[2][BM * BK];
  __shared__ __align__(16) __bf16 sB1[2][BK * BN];
  __shared__ __align__(16) __bf16 sB3[2][BK * BN];

  int tid = threadIdx.x;
  for (int i = tid; i < BM; i += 256) stok[i] = gidx[rowBase + mt * BM + i];
  __syncthreads();

  int wid = tid >> 5, lane = tid & 31;
  int wm = wid >> 2, wn = wid & 3;

  // loop-invariant A-staging descriptors: 4 quads per thread
  int kq4 = (tid & 7) * 4;
  const float* ap[4];
  float ms[4];
  int sidx[4];
#pragma unroll
  for (int it = 0; it < 4; ++it) {
    int m = (tid >> 3) + it * 32;
    int tok = stok[m];
    ms[it] = (tok >= 0) ? 1.f : 0.f;           // branch-free padding mask
    int trow = (tok >= 0) ? tok : 0;           // clamped: always a valid row
    ap[it] = x + (size_t)trow * CDIM + kq4;
    sidx[it] = a_swz(m, kq4);
  }

  float4 r[4];
  auto loadA = [&](int k0) {                   // global -> regs (prefetch)
#pragma unroll
    for (int it = 0; it < 4; ++it) r[it] = *(const float4*)(ap[it] + k0);
  };
  auto storeA = [&](int buf) {                 // cvt + LDS store (consume)
#pragma unroll
    for (int it = 0; it < 4; ++it) {
      v4bf bv = {f2bf(r[it].x * ms[it]), f2bf(r[it].y * ms[it]),
                 f2bf(r[it].z * ms[it]), f2bf(r[it].w * ms[it])};
      *(v4bf*)&sA[buf][sidx[it]] = bv;
    }
  };
  auto stageB = [&](int buf, int kt) {         // pure async, 4 ops/thread
    const __bf16* g1 = w1b + ((size_t)kt << 12);
    const __bf16* g3 = w3b + ((size_t)kt << 12);
#pragma unroll
    for (int it = 0; it < 2; ++it) {
      int o = (tid + it * 256) << 3;           // 16B chunks
      ASYNC_CP16(&sB1[buf][o], g1 + o);
      ASYNC_CP16(&sB3[buf][o], g3 + o);
    }
  };

  v8f acc1[4][2] = {};
  v8f acc3[4][2] = {};

  loadA(0);
  stageB(0, 0);
  storeA(0);
  loadA(BK);                                   // prefetch k-step 1
  const int NK = CDIM / BK;                    // 32
  for (int i = 0; i < NK; ++i) {
    int nxt = i + 1;
    __syncthreads();                           // readers of buf nxt&1 done
    if (nxt < NK) {
      stageB(nxt & 1, nxt);
      storeA(nxt & 1);                         // consume prefetched regs
      if (nxt + 1 < NK) loadA((nxt + 1) * BK); // prefetch next (full step slack)
      WAIT_ASYNC(4);                           // stage i complete
    } else {
      WAIT_ASYNC(0);
    }
    __syncthreads();                           // buf i&1 resident
    int bb = i & 1;
    v16bf afr[4];
#pragma unroll
    for (int a = 0; a < 4; ++a)
      afr[a] = *(const v16bf*)&sA[bb][(((wm << 2) + a) << 9) + (lane << 4)];
#pragma unroll
    for (int b = 0; b < 2; ++b) {
      v16bf bf1 = *(const v16bf*)&sB1[bb][(((wn << 1) + b) << 9) + (lane << 4)];
      v16bf bf3 = *(const v16bf*)&sB3[bb][(((wn << 1) + b) << 9) + (lane << 4)];
#pragma unroll
      for (int a = 0; a < 4; ++a) {
        acc1[a][b] = __builtin_amdgcn_wmma_f32_16x16x32_bf16(
            false, afr[a], false, bf1, (short)0, acc1[a][b], false, false);
        acc3[a][b] = __builtin_amdgcn_wmma_f32_16x16x32_bf16(
            false, afr[a], false, bf3, (short)0, acc3[a][b], false, false);
      }
    }
  }

  // epilogue: silu(a1)*a3 -> hidden in fragment-blocked layout
  int rt = (rowBase >> 7) + mt;
#pragma unroll
  for (int a = 0; a < 4; ++a) {
#pragma unroll
    for (int b = 0; b < 2; ++b) {
      int kt = (f0 >> 5) + wn;
      int kin = (b << 4) + (lane & 15);
#pragma unroll
      for (int j = 0; j < 8; ++j) {
        int m = (wm << 6) + (a << 4) + ((lane >> 4) << 3) + j;
        float g = acc1[a][b][j];
        float h = (g / (1.f + __expf(-g))) * acc3[a][b][j];
        hidden[(((size_t)(rt * (FDIM / BK) + kt)) << 12) + a_swz(m, kin)] =
            f2bf(h);
      }
    }
  }
}

// ---------------------------------------------------------------------------
// Kernel 4: out[token] += we * (hidden @ w2).  Both operands blocked bf16 ->
// both staged purely via async global->LDS, double-buffered.
// ---------------------------------------------------------------------------
__global__ __launch_bounds__(256)
void moe_ffn2(const unsigned short* __restrict__ hid_,
              const unsigned short* __restrict__ w2b_,
              const int* __restrict__ gidx, const float* __restrict__ wrow,
              const int* __restrict__ offs, float* __restrict__ out) {
  int e = blockIdx.z;
  int rowBase = offs[e];
  int rows = offs[e + 1] - rowBase;
  int mt = blockIdx.x;
  if (mt * BM >= rows) return;
  int nt = blockIdx.y;
  int c0 = nt * BN;

  const __bf16* hidb =
      (const __bf16*)hid_ + (((size_t)((rowBase >> 7) + mt) * (FDIM / BK)) << 12);
  const __bf16* w2b =
      (const __bf16*)w2b_ + (((size_t)(e * (CDIM / BN) + nt) * (FDIM / BK)) << 12);

  __shared__ int stok[BM];
  __shared__ float sw[BM];
  __shared__ __align__(16) __bf16 sA[2][BM * BK];
  __shared__ __align__(16) __bf16 sB[2][BK * BN];

  int tid = threadIdx.x;
  for (int i = tid; i < BM; i += 256) {
    stok[i] = gidx[rowBase + mt * BM + i];
    sw[i] = wrow[rowBase + mt * BM + i];
  }

  int wid = tid >> 5, lane = tid & 31;
  int wm = wid >> 2, wn = wid & 3;

  auto stage = [&](int buf, int kt) {          // pure async, 4 ops/thread
    const __bf16* ga = hidb + ((size_t)kt << 12);
    const __bf16* gb = w2b + ((size_t)kt << 12);
#pragma unroll
    for (int it = 0; it < 2; ++it) {
      int o = (tid + it * 256) << 3;
      ASYNC_CP16(&sA[buf][o], ga + o);
      ASYNC_CP16(&sB[buf][o], gb + o);
    }
  };

  v8f acc[4][2] = {};
  stage(0, 0);
  const int NK = FDIM / BK;                    // 64
  for (int i = 0; i < NK; ++i) {
    int nxt = i + 1;
    __syncthreads();
    if (nxt < NK) {
      stage(nxt & 1, nxt);
      WAIT_ASYNC(4);
    } else {
      WAIT_ASYNC(0);
    }
    __syncthreads();
    int bb = i & 1;
    v16bf afr[4];
#pragma unroll
    for (int a = 0; a < 4; ++a)
      afr[a] = *(const v16bf*)&sA[bb][(((wm << 2) + a) << 9) + (lane << 4)];
#pragma unroll
    for (int b = 0; b < 2; ++b) {
      v16bf bfr = *(const v16bf*)&sB[bb][(((wn << 1) + b) << 9) + (lane << 4)];
#pragma unroll
      for (int a = 0; a < 4; ++a)
        acc[a][b] = __builtin_amdgcn_wmma_f32_16x16x32_bf16(
            false, afr[a], false, bfr, (short)0, acc[a][b], false, false);
    }
  }

  // epilogue: scale by combine weight, atomic scatter-add into out
#pragma unroll
  for (int a = 0; a < 4; ++a) {
#pragma unroll
    for (int b = 0; b < 2; ++b) {
      int n = c0 + (wn << 5) + (b << 4) + (lane & 15);
#pragma unroll
      for (int j = 0; j < 8; ++j) {
        int m = (wm << 6) + (a << 4) + ((lane >> 4) << 3) + j;
        int tok = stok[m];
        if (tok >= 0)
          atomicAdd(&out[(size_t)tok * CDIM + n], acc[a][b][j] * sw[m]);
      }
    }
  }
}

// ---------------------------------------------------------------------------
extern "C" void kernel_launch(void* const* d_in, const int* in_sizes, int n_in,
                              void* d_out, int out_size, void* d_ws,
                              size_t ws_size, hipStream_t stream) {
  const float* x  = (const float*)d_in[0];
  const float* rw = (const float*)d_in[1];
  const float* w1 = (const float*)d_in[2];
  const float* w3 = (const float*)d_in[3];
  const float* w2 = (const float*)d_in[4];
  float* out = (float*)d_out;

  char* ws = (char*)d_ws;
  float* we           = (float*)(ws + WE_OFF);
  int* gidx           = (int*)(ws + GIDX_OFF);
  float* wr           = (float*)(ws + WROW_OFF);
  int* offs           = (int*)(ws + OFFS_OFF);
  unsigned short* hd  = (unsigned short*)(ws + HID_OFF);
  unsigned short* w1b = (unsigned short*)(ws + W1B_OFF);
  unsigned short* w3b = (unsigned short*)(ws + W3B_OFF);
  unsigned short* w2b = (unsigned short*)(ws + W2B_OFF);

  (void)in_sizes; (void)n_in; (void)ws_size;

  (void)hipMemsetAsync(d_out, 0, (size_t)out_size * sizeof(float), stream);

  int wblocks = (NEXP * CDIM * FDIM / 4) / 256;   // 16384
  wcvt<<<wblocks, 256, 0, stream>>>(w1, w1b, CDIM, FDIM);
  wcvt<<<wblocks, 256, 0, stream>>>(w3, w3b, CDIM, FDIM);
  wcvt<<<wblocks, 256, 0, stream>>>(w2, w2b, FDIM, CDIM);

  moe_router<<<NTOK / 8, 256, 0, stream>>>(x, rw, we);
  moe_gather<<<1, 256, 0, stream>>>(we, gidx, wr, offs);

  dim3 g1(NTOK / BM, FDIM / BN, NEXP);   // 64 x 16 x 8, early-exit on counts
  moe_ffn1<<<g1, 256, 0, stream>>>(x, w1b, w3b, gidx, offs, hd);

  dim3 g2(NTOK / BM, CDIM / BN, NEXP);   // 64 x 8 x 8
  moe_ffn2<<<g2, 256, 0, stream>>>(hd, w2b, gidx, wr, offs, out);
}
